// CTC_76587856822431
// MI455X (gfx1250) — matrix operand order repaired and checked by reading, not compiled, not run on hardware
//
#include <hip/hip_runtime.h>

// Problem constants (from reference)
#define Bn    16
#define Tn    512
#define Dn    256
#define Vn    4233
#define LMAXn 64
#define Sn    (2 * LMAXn + 1)   // 129
#define NEGF  (-1e30f)
#define VTILES ((Vn + 15) / 16) // 265
#define MROWS 32                // rows per workgroup (2 x 16-row WMMA tiles)

typedef __attribute__((ext_vector_type(16))) __bf16 v16bf;
typedef __attribute__((ext_vector_type(8)))  __bf16 v8bf;
typedef __attribute__((ext_vector_type(8)))  float  v8f;
typedef __attribute__((ext_vector_type(4)))  unsigned int v4u;
typedef __attribute__((ext_vector_type(8)))  int    v8i;
typedef __attribute__((ext_vector_type(4)))  int    v4i;

#if __has_builtin(__builtin_amdgcn_tensor_load_to_lds)
#define HAVE_TDM 1
#else
#define HAVE_TDM 0
#endif

// ---------------------------------------------------------------------------
// K0/K1: fp32 -> bf16 conversion (used for both hs and W)
// ---------------------------------------------------------------------------
__global__ void cvt_bf16_kernel(const float* __restrict__ in,
                                __bf16* __restrict__ out, int n) {
  int i = blockIdx.x * blockDim.x + threadIdx.x;
  if (i < n) out[i] = (__bf16)in[i];
}

__global__ void zero_out_kernel(float* out) { out[0] = 0.0f; }

// ---------------------------------------------------------------------------
// K2: fused WMMA-bf16 GEMM (hs @ W^T + b) with streaming logsumexp over V.
// grid = (B*T)/32 workgroups of 256 threads (8 wave32).
// 32x256 bf16 A tile staged to LDS by the Tensor Data Mover (padded rows);
// each wave sweeps V tiles with B fragments double-buffered in VGPRs; every
// B fragment feeds TWO WMMAs (M-blocking) to halve W panel traffic.
// ---------------------------------------------------------------------------
__global__ __launch_bounds__(256) void gemm_lse_kernel(
    const __bf16* __restrict__ Ab,    // [B*T, D] bf16
    const __bf16* __restrict__ Wb,    // [V, D]  bf16
    const float*  __restrict__ bias,  // [V]
    float*        __restrict__ lse)   // [B*T]
{
  __shared__ __bf16 As[MROWS][Dn + 8]; // 528B row stride (TDM pad) -> no bank conflicts
  __shared__ float redm[8][MROWS];
  __shared__ float reds[8][MROWS];

  const int tid  = threadIdx.x;
  const int wave = tid >> 5;
  const int lane = tid & 31;
  const int row0 = blockIdx.x * MROWS;

#if HAVE_TDM
  // --- Tensor Data Mover: DMA the 32x256 bf16 A tile into LDS with padding ---
  if (wave == 0) {
    unsigned long long gaddr =
        (unsigned long long)(uintptr_t)(Ab + (size_t)row0 * Dn);
    unsigned int laddr = (unsigned int)(uintptr_t)&As[0][0]; // low 32b of flat = LDS offset

    v4u g0;
    g0[0] = 1u;                                   // count=1 valid user descriptor
    g0[1] = laddr;                                // lds_addr
    g0[2] = (unsigned int)gaddr;                  // global_addr[31:0]
    g0[3] = (unsigned int)((gaddr >> 32) & 0x1FFFFFFu) | (2u << 30); // [56:32] | type=2

    v8i g1;
    // data_size=1 (2B) | pad_enable | pad_interval=6 (128 dw = 512B) | pad_amount=3 (4 dw = 16B)
    g1[0] = (int)((1u << 16) | (1u << 20) | (6u << 22) | (3u << 25));
    g1[1] = (int)(((unsigned)Dn & 0xFFFFu) << 16);               // tensor_dim0[15:0]=256
    g1[2] = (int)(((unsigned)Dn >> 16) | ((unsigned)MROWS << 16)); // dim0 hi | tensor_dim1=32
    g1[3] = (int)(((unsigned)Dn & 0xFFFFu) << 16);               // tensor_dim1 hi=0 | tile_dim0=256
    g1[4] = (int)((unsigned)MROWS);                              // tile_dim1=32 | tile_dim2=0
    g1[5] = (int)Dn;                                             // tensor_dim0_stride=256
    g1[6] = 0;
    g1[7] = 0;

    v4i gz4 = {0, 0, 0, 0};                       // 2-D tensor: groups 2/3 unused
    v8i gz8 = {0, 0, 0, 0, 0, 0, 0, 0};
    __builtin_amdgcn_tensor_load_to_lds(g0, g1, gz4, gz4, gz8, 0);
    __builtin_amdgcn_s_wait_tensorcnt(0);
  }
#else
  // Fallback: cooperative bf16 copy into LDS (32 rows x 256 cols)
  {
    const int r  = tid >> 3;
    const int c0 = (tid & 7) * 32;
    const __bf16* src = Ab + (size_t)(row0 + r) * Dn + c0;
    #pragma unroll
    for (int j = 0; j < 32; ++j) As[r][c0 + j] = src[j];
  }
#endif
  __syncthreads();

  const int n      = lane & 15;       // B/C column within tile
  const int hi     = lane >> 4;       // lane half
  const int abase  = hi * 16;         // A-fragment K sub-base (contiguous, matches B)
  const int bbase  = hi * 16;         // B-fragment K sub-base
  const int rowsel = hi * 8;          // C rows handled by this lane (within 16-row tile)

  // Hoist A fragments for both 16-row sub-tiles into registers.
  v16bf afrag[2][8];
  #pragma unroll
  for (int u = 0; u < 2; ++u) {
    const int m = u * 16 + (lane & 15);
    #pragma unroll
    for (int kc = 0; kc < 8; ++kc) {
      const __bf16* ap = &As[m][kc * 32 + abase];
      v8bf alo = *(const v8bf*)(ap);
      v8bf ahi = *(const v8bf*)(ap + 8);
      afrag[u][kc] = __builtin_shufflevector(alo, ahi,
          0, 1, 2, 3, 4, 5, 6, 7, 8, 9, 10, 11, 12, 13, 14, 15);
    }
  }

  float mrun[2][8], srun[2][8];
  #pragma unroll
  for (int u = 0; u < 2; ++u)
    #pragma unroll
    for (int r = 0; r < 8; ++r) { mrun[u][r] = NEGF; srun[u][r] = 0.0f; }

  // Double-buffered B fragments: load next V-tile while WMMAing current one.
  v16bf bcur[8], bnxt[8];
  {
    const int vcol  = wave * 16 + n;
    const __bf16* wrow = Wb + (size_t)(vcol < Vn ? vcol : Vn - 1) * Dn + bbase;
    #pragma unroll
    for (int kc = 0; kc < 8; ++kc) bcur[kc] = *(const v16bf*)(wrow + kc * 32);
  }

  for (int ct = wave; ct < VTILES; ct += 8) {
    if (ct + 8 < VTILES) {
      const int vcol  = (ct + 8) * 16 + n;
      const __bf16* wrow = Wb + (size_t)(vcol < Vn ? vcol : Vn - 1) * Dn + bbase;
      #pragma unroll
      for (int kc = 0; kc < 8; ++kc) bnxt[kc] = *(const v16bf*)(wrow + kc * 32);
    }

    v8f acc0 = {}, acc1 = {};
    #pragma unroll
    for (int kc = 0; kc < 8; ++kc) {
      acc0 = __builtin_amdgcn_wmma_f32_16x16x32_bf16(
          false, afrag[0][kc], false, bcur[kc], (short)0, acc0, false, false);
      acc1 = __builtin_amdgcn_wmma_f32_16x16x32_bf16(
          false, afrag[1][kc], false, bcur[kc], (short)0, acc1, false, false);
    }

    const int  vcol   = ct * 16 + n;
    const bool valid  = vcol < Vn;
    const float bias_v = valid ? bias[vcol] : 0.0f;
    #pragma unroll
    for (int r = 0; r < 8; ++r) {
      float x0 = valid ? (acc0[r] + bias_v) : NEGF;
      float nm0 = fmaxf(mrun[0][r], x0);
      srun[0][r] = srun[0][r] * __expf(mrun[0][r] - nm0) + __expf(x0 - nm0);
      mrun[0][r] = nm0;
      float x1 = valid ? (acc1[r] + bias_v) : NEGF;
      float nm1 = fmaxf(mrun[1][r], x1);
      srun[1][r] = srun[1][r] * __expf(mrun[1][r] - nm1) + __expf(x1 - nm1);
      mrun[1][r] = nm1;
    }

    #pragma unroll
    for (int kc = 0; kc < 8; ++kc) bcur[kc] = bnxt[kc];
  }

  // Reduce (max,sumexp) across the 16 lanes of each half (columns of this wave)
  #pragma unroll
  for (int mask = 8; mask >= 1; mask >>= 1) {
    #pragma unroll
    for (int u = 0; u < 2; ++u)
      #pragma unroll
      for (int r = 0; r < 8; ++r) {
        float om = __shfl_xor(mrun[u][r], mask, 32);
        float os = __shfl_xor(srun[u][r], mask, 32);
        float nm = fmaxf(mrun[u][r], om);
        srun[u][r] = srun[u][r] * __expf(mrun[u][r] - nm) + os * __expf(om - nm);
        mrun[u][r] = nm;
      }
  }
  if (n == 0) {
    #pragma unroll
    for (int u = 0; u < 2; ++u)
      #pragma unroll
      for (int r = 0; r < 8; ++r) {
        redm[wave][u * 16 + rowsel + r] = mrun[u][r];
        reds[wave][u * 16 + rowsel + r] = srun[u][r];
      }
  }
  __syncthreads();

  // Combine the 8 waves' partials; one thread per row.
  if (tid < MROWS) {
    float mm = NEGF, ss = 0.0f;
    #pragma unroll
    for (int w = 0; w < 8; ++w) {
      float om = redm[w][tid], os = reds[w][tid];
      float nm = fmaxf(mm, om);
      ss = ss * __expf(mm - nm) + os * __expf(om - nm);
      mm = nm;
    }
    lse[row0 + tid] = mm + __logf(ss);
  }
}

// ---------------------------------------------------------------------------
// K3: per-(b,t) gather of the S=129 extended-label log-probs.
// lp[b,t,s] = <hs[b,t], W[ext_s]> + bias[ext_s] - lse[b,t]
// ---------------------------------------------------------------------------
__global__ __launch_bounds__(256) void label_logp_kernel(
    const float* __restrict__ hs,
    const int*   __restrict__ ys_pad,
    const float* __restrict__ W,
    const float* __restrict__ bias,
    const float* __restrict__ lse,
    float*       __restrict__ lp)
{
  __shared__ float hrow[Dn];
  const int row = blockIdx.x;            // b*T + t
  const int b   = row / Tn;
  const int tid = threadIdx.x;
  hrow[tid] = hs[(size_t)row * Dn + tid];
  __syncthreads();

  const int wave = tid >> 5, lane = tid & 31;
  const float myLse = lse[row];
  for (int s = wave; s < Sn; s += 8) {
    const int v = (s & 1) ? ys_pad[b * LMAXn + (s >> 1)] : 0;   // blank = 0
    const float* wr = W + (size_t)v * Dn;
    float acc = 0.0f;
    #pragma unroll
    for (int d = lane; d < Dn; d += 32) acc += hrow[d] * wr[d];
    #pragma unroll
    for (int mask = 16; mask >= 1; mask >>= 1) acc += __shfl_xor(acc, mask, 32);
    if (lane == 0) lp[(size_t)row * Sn + s] = acc + bias[v] - myLse;
  }
}

// ---------------------------------------------------------------------------
// K4: CTC alpha recursion. One workgroup per batch element; state s = tid.
// ---------------------------------------------------------------------------
__global__ __launch_bounds__(256) void ctc_alpha_kernel(
    const float* __restrict__ lp,      // [B, T, S]
    const int*   __restrict__ ys_pad,
    const int*   __restrict__ ys_lens,
    const int*   __restrict__ hlens,
    float*       __restrict__ out)
{
  __shared__ float alpha[Sn];
  const int b  = blockIdx.x;
  const int s  = threadIdx.x;
  const int hl = hlens[b];

  int ext_s  = 0, ext_s2 = 0;
  if (s < Sn) {
    ext_s  = (s & 1) ? ys_pad[b * LMAXn + (s >> 1)] : 0;
    ext_s2 = (s >= 2 && ((s - 2) & 1)) ? ys_pad[b * LMAXn + ((s - 2) >> 1)] : 0;
  }
  const bool allow = (s >= 2) && (ext_s != 0) && (ext_s != ext_s2);
  const float* lpb = lp + (size_t)b * Tn * Sn;

  if (s < Sn) alpha[s] = (s < 2) ? lpb[s] : NEGF;
  __syncthreads();

  for (int t = 1; t < Tn; ++t) {
    float a0 = NEGF, a1 = NEGF, a2 = NEGF;
    if (s < Sn) {
      a0 = alpha[s];
      if (s >= 1) a1 = alpha[s - 1];
      if (allow)  a2 = alpha[s - 2];
    }
    __syncthreads();
    if (s < Sn && t < hl) {
      float m = fmaxf(a0, fmaxf(a1, a2));
      alpha[s] = m + __logf(__expf(a0 - m) + __expf(a1 - m) + __expf(a2 - m))
               + lpb[t * Sn + s];
    }
    __syncthreads();
  }

  if (s == 0) {
    const int last = 2 * ys_lens[b];
    float x = alpha[last], y = alpha[last - 1];
    float m = fmaxf(x, y);
    float ll = m + __logf(__expf(x - m) + __expf(y - m));
    atomicAdd(out, -ll / (float)Bn);
  }
}

// ---------------------------------------------------------------------------
// Launch
// ---------------------------------------------------------------------------
extern "C" void kernel_launch(void* const* d_in, const int* in_sizes, int n_in,
                              void* d_out, int out_size, void* d_ws, size_t ws_size,
                              hipStream_t stream) {
  const float* hs      = (const float*)d_in[0];  // [B,T,D]
  const int*   hlens   = (const int*)  d_in[1];  // [B]
  const int*   ys_pad  = (const int*)  d_in[2];  // [B,LMAX]
  const int*   ys_lens = (const int*)  d_in[3];  // [B]
  const float* W       = (const float*)d_in[4];  // [V,D]
  const float* bias    = (const float*)d_in[5];  // [V]
  float* out = (float*)d_out;

  // Workspace layout (256B-aligned offsets). Total ~10.7 MB.
  char* ws = (char*)d_ws;
  size_t szAb   = (size_t)Bn * Tn * Dn * sizeof(__bf16);    // 4,194,304
  size_t offWb  = (szAb + 255) & ~(size_t)255;
  size_t szWb   = (size_t)Vn * Dn * sizeof(__bf16);         // 2,167,296
  size_t offLse = (offWb + szWb + 255) & ~(size_t)255;
  size_t szLse  = (size_t)Bn * Tn * sizeof(float);
  size_t offLp  = (offLse + szLse + 255) & ~(size_t)255;    // lp: B*T*S*4

  __bf16* Ab  = (__bf16*)ws;
  __bf16* Wb  = (__bf16*)(ws + offWb);
  float*  lse = (float*)(ws + offLse);
  float*  lp  = (float*)(ws + offLp);

  zero_out_kernel<<<1, 1, 0, stream>>>(out);
  cvt_bf16_kernel<<<(Bn * Tn * Dn + 255) / 256, 256, 0, stream>>>(hs, Ab, Bn * Tn * Dn);
  cvt_bf16_kernel<<<(Vn * Dn + 255) / 256, 256, 0, stream>>>(W, Wb, Vn * Dn);
  gemm_lse_kernel<<<(Bn * Tn) / MROWS, 256, 0, stream>>>(Ab, Wb, bias, lse);
  label_logp_kernel<<<Bn * Tn, 256, 0, stream>>>(hs, ys_pad, W, bias, lse, lp);
  ctc_alpha_kernel<<<Bn, 256, 0, stream>>>(lp, ys_pad, ys_lens, hlens, out);
}
